// GeometryAwareCrossAttentionBlock_67242007986300
// MI455X (gfx1250) — compile-verified, matrix operand order checked by reading, not run on hardware
//
#include <hip/hip_runtime.h>
#include <cstdint>
#include <cstddef>

#define BATCH  4
#define NQPTS  2048
#define NKPTS  2048
#define DMODEL 384
#define NHEAD  6
#define DHEAD  64
#define MROWS  (BATCH * NQPTS)
#define KNN    8
#define CTOT   (3 + DMODEL)   // 387

#if defined(__has_builtin)
#if __has_builtin(__builtin_amdgcn_global_load_async_to_lds_b128)
#define HAVE_ASYNC_LDS 1
#endif
#endif
#ifndef HAVE_ASYNC_LDS
#define HAVE_ASYNC_LDS 0
#endif

typedef __attribute__((ext_vector_type(16))) __bf16 v16bf;
typedef __attribute__((ext_vector_type(8)))  float  v8f;
typedef int v4i_ __attribute__((vector_size(16)));   // matches builtin param type
typedef __attribute__((address_space(1))) v4i_* gas_v4i;   // global int4*
typedef __attribute__((address_space(3))) v4i_* las_v4i;   // LDS int4*

union Frag { unsigned int u[8]; v16bf v; };

static __device__ __forceinline__ v8f wmma_bf16(const Frag& a, const Frag& b, v8f c) {
  return __builtin_amdgcn_wmma_f32_16x16x32_bf16(false, a.v, false, b.v, (short)0, c, false, false);
}

static __device__ __forceinline__ unsigned short f2bf(float f) {
  unsigned int u = __float_as_uint(f);
  u += 0x7FFFu + ((u >> 16) & 1u);          // round-to-nearest-even
  return (unsigned short)(u >> 16);
}

static __device__ __forceinline__ v8f zero8() {
  v8f z;
  #pragma unroll
  for (int i = 0; i < 8; ++i) z[i] = 0.f;
  return z;
}

static __device__ __forceinline__ void wait_async0() {
#if defined(__has_builtin)
#if __has_builtin(__builtin_amdgcn_s_wait_asynccnt)
  __builtin_amdgcn_s_wait_asynccnt(0);
  return;
#else
  asm volatile("s_wait_asynccnt 0x0" ::: "memory");
  return;
#endif
#else
  asm volatile("s_wait_asynccnt 0x0" ::: "memory");
#endif
}

// K-pair index inside a 16x32 bf16 A (or mirrored B) fragment, per ISA 7.12.2
#define KPAIR(r, lh) (2 * ((r) & 3) + (lh) * 8 + ((r) >> 2) * 16)

// ---------------------------------------------------------------------------
// Weight prep: fp32 W[(k+r0)][n] (row stride ldw) -> bf16 Wt[n][k], n-major
// ---------------------------------------------------------------------------
__global__ __launch_bounds__(256) void wprep_kernel(const float* __restrict__ W,
                                                    unsigned short* __restrict__ Wt,
                                                    int Kd, int N, int r0, int ldw) {
  int t = blockIdx.x * 256 + threadIdx.x;
  if (t >= N * Kd) return;
  int n = t / Kd, k = t - n * Kd;
  Wt[t] = f2bf(W[(size_t)(k + r0) * ldw + n]);
}

// ---------------------------------------------------------------------------
// Split (B, 3+D, N) -> coords (B,N,3) + feats (B,N,D)
// ---------------------------------------------------------------------------
__global__ __launch_bounds__(256) void split_kernel(const float* __restrict__ pts,
                                                    float* __restrict__ coords,
                                                    float* __restrict__ feats, int Nn) {
  size_t t = (size_t)blockIdx.x * 256 + threadIdx.x;
  size_t total = (size_t)BATCH * Nn * CTOT;
  if (t >= total) return;
  int b = (int)(t / ((size_t)Nn * CTOT));
  int r = (int)(t - (size_t)b * Nn * CTOT);
  int n = r / CTOT, c = r - n * CTOT;
  float v = pts[(size_t)b * CTOT * Nn + (size_t)c * Nn + n];
  if (c < 3) coords[((size_t)b * Nn + n) * 3 + c] = v;
  else       feats [((size_t)b * Nn + n) * DMODEL + (c - 3)] = v;
}

// ---------------------------------------------------------------------------
// Pack coords + feats -> (B, 3+D, N)
// ---------------------------------------------------------------------------
__global__ __launch_bounds__(256) void pack_kernel(const float* __restrict__ coords,
                                                   const float* __restrict__ feats,
                                                   float* __restrict__ out, int Nn) {
  size_t t = (size_t)blockIdx.x * 256 + threadIdx.x;
  size_t total = (size_t)BATCH * Nn * CTOT;
  if (t >= total) return;
  int b = (int)(t / ((size_t)Nn * CTOT));
  int r = (int)(t - (size_t)b * Nn * CTOT);
  int c = r / Nn, n = r - c * Nn;
  float v = (c < 3) ? coords[((size_t)b * Nn + n) * 3 + c]
                    : feats [((size_t)b * Nn + n) * DMODEL + (c - 3)];
  out[t] = v;
}

// ---------------------------------------------------------------------------
// LayerNorm (D=384), one wave per row, bf16 output
// ---------------------------------------------------------------------------
__global__ __launch_bounds__(256) void ln_bf16_kernel(const float* __restrict__ x,
                                                      const float* __restrict__ gam,
                                                      const float* __restrict__ bet,
                                                      unsigned short* __restrict__ y) {
  int wave = threadIdx.x >> 5, lane = threadIdx.x & 31;
  int row = blockIdx.x * 8 + wave;
  const float* xr = x + (size_t)row * DMODEL;
  float v[12]; float s = 0.f;
  #pragma unroll
  for (int i = 0; i < 12; ++i) { v[i] = xr[lane + i * 32]; s += v[i]; }
  #pragma unroll
  for (int o = 1; o < 32; o <<= 1) s += __shfl_xor(s, o, 32);
  float mean = s * (1.f / DMODEL);
  float q = 0.f;
  #pragma unroll
  for (int i = 0; i < 12; ++i) { float d = v[i] - mean; q += d * d; }
  #pragma unroll
  for (int o = 1; o < 32; o <<= 1) q += __shfl_xor(q, o, 32);
  float rstd = rsqrtf(q * (1.f / DMODEL) + 1e-5f);
  unsigned short* yr = y + (size_t)row * DMODEL;
  #pragma unroll
  for (int i = 0; i < 12; ++i) {
    int c = lane + i * 32;
    yr[c] = f2bf((v[i] - mean) * rstd * gam[c] + bet[c]);
  }
}

// ---------------------------------------------------------------------------
// bf16 WMMA GEMM: out = act(A[M,Kd] @ Wt[N,Kd]^T + bias) (+ residual)
// 4 waves, 128x64 block tile, K-step 32, double-buffered LDS software
// pipeline. Each wave: 2 A-frags x 4 B-frags = 8 WMMA per K-step.
// Tiles move global->LDS via GLOBAL_LOAD_ASYNC_TO_LDS_B128 (ASYNCcnt DMA,
// no VGPR staging) when the toolchain exposes it; register-staged fallback
// otherwise. LDS ld = 40 elems (80B rows: 16B aligned, conflict-free reads).
// ---------------------------------------------------------------------------
__global__ __launch_bounds__(128, 1) void gemm_bf16_kernel(
    const unsigned short* __restrict__ A, int lda,
    const unsigned short* __restrict__ Wt,
    const float* __restrict__ bias,
    const float* __restrict__ residual, int ldr,
    float* __restrict__ outF, int ldof,
    unsigned short* __restrict__ outB, int ldob, int obcol,
    int Kd, int act) {
  __shared__ unsigned short sA[2][128 * 40];
  __shared__ unsigned short sB[2][64 * 40];
  const int tid = threadIdx.x;
  const int wave = tid >> 5, lane = tid & 31;
  const int lm = lane & 15, lh = lane >> 4;
  const int mrow0 = blockIdx.x * 128;
  const int ncol0 = blockIdx.y * 64;

  v8f acc[2][4];
  #pragma unroll
  for (int hf = 0; hf < 2; ++hf)
    #pragma unroll
    for (int nt = 0; nt < 4; ++nt) acc[hf][nt] = zero8();

  const int nkt = Kd >> 5;

#if HAVE_ASYNC_LDS
  auto issue_tile = [&](int buf, int k0) {
    #pragma unroll
    for (int i = 0; i < 4; ++i) {                 // A tile 128x32
      int c = tid + i * 128, row = c >> 2, col8 = (c & 3) * 8;
      __builtin_amdgcn_global_load_async_to_lds_b128(
          (gas_v4i)(A + (size_t)(mrow0 + row) * lda + k0 + col8),
          (las_v4i)(&sA[buf][row * 40 + col8]), 0, 0);
    }
    #pragma unroll
    for (int i = 0; i < 2; ++i) {                 // B tile 64x32 (n-major)
      int c = tid + i * 128, row = c >> 2, col8 = (c & 3) * 8;
      __builtin_amdgcn_global_load_async_to_lds_b128(
          (gas_v4i)(Wt + (size_t)(ncol0 + row) * Kd + k0 + col8),
          (las_v4i)(&sB[buf][row * 40 + col8]), 0, 0);
    }
  };
  issue_tile(0, 0);
#else
  uint4 ra[4], rb[2];
  // prologue: load + stage K-step 0 into buffer 0
  #pragma unroll
  for (int i = 0; i < 4; ++i) {
    int c = tid + i * 128, row = c >> 2, col8 = (c & 3) * 8;
    ra[i] = *(const uint4*)(A + (size_t)(mrow0 + row) * lda + col8);
  }
  #pragma unroll
  for (int i = 0; i < 2; ++i) {
    int c = tid + i * 128, row = c >> 2, col8 = (c & 3) * 8;
    rb[i] = *(const uint4*)(Wt + (size_t)(ncol0 + row) * Kd + col8);
  }
  #pragma unroll
  for (int i = 0; i < 4; ++i) {
    int c = tid + i * 128, row = c >> 2, col8 = (c & 3) * 8;
    *(uint4*)(&sA[0][row * 40 + col8]) = ra[i];
  }
  #pragma unroll
  for (int i = 0; i < 2; ++i) {
    int c = tid + i * 128, row = c >> 2, col8 = (c & 3) * 8;
    *(uint4*)(&sB[0][row * 40 + col8]) = rb[i];
  }
#endif

  for (int kt = 0; kt < nkt; ++kt) {
    const int cur = kt & 1, nxt = cur ^ 1;
    const bool more = (kt + 1 < nkt);

#if HAVE_ASYNC_LDS
    wait_async0();                                     // my cur-tile DMA done
    __syncthreads();                                   // everyone's done; nxt free
    if (more) issue_tile(nxt, (kt + 1) << 5);          // DMA overlaps WMMA below
#else
    if (more) {                                        // issue next K-step's globals
      int k0 = (kt + 1) << 5;
      #pragma unroll
      for (int i = 0; i < 4; ++i) {
        int c = tid + i * 128, row = c >> 2, col8 = (c & 3) * 8;
        ra[i] = *(const uint4*)(A + (size_t)(mrow0 + row) * lda + k0 + col8);
      }
      #pragma unroll
      for (int i = 0; i < 2; ++i) {
        int c = tid + i * 128, row = c >> 2, col8 = (c & 3) * 8;
        rb[i] = *(const uint4*)(Wt + (size_t)(ncol0 + row) * Kd + k0 + col8);
      }
    }
    __syncthreads();
#endif

    Frag a0, a1;
    #pragma unroll
    for (int r = 0; r < 8; ++r) {
      a0.u[r] = *(const unsigned int*)(&sA[cur][(wave * 32 + lm)      * 40 + KPAIR(r, lh)]);
      a1.u[r] = *(const unsigned int*)(&sA[cur][(wave * 32 + 16 + lm) * 40 + KPAIR(r, lh)]);
    }
    #pragma unroll
    for (int nt = 0; nt < 4; ++nt) {
      Frag bf;
      #pragma unroll
      for (int r = 0; r < 8; ++r)
        bf.u[r] = *(const unsigned int*)(&sB[cur][(nt * 16 + lm) * 40 + KPAIR(r, lh)]);
      acc[0][nt] = wmma_bf16(a0, bf, acc[0][nt]);
      acc[1][nt] = wmma_bf16(a1, bf, acc[1][nt]);
    }

#if !HAVE_ASYNC_LDS
    if (more) {                                        // stage next into other buffer
      #pragma unroll
      for (int i = 0; i < 4; ++i) {
        int c = tid + i * 128, row = c >> 2, col8 = (c & 3) * 8;
        *(uint4*)(&sA[nxt][row * 40 + col8]) = ra[i];
      }
      #pragma unroll
      for (int i = 0; i < 2; ++i) {
        int c = tid + i * 128, row = c >> 2, col8 = (c & 3) * 8;
        *(uint4*)(&sB[nxt][row * 40 + col8]) = rb[i];
      }
    }
#endif
  }

  #pragma unroll
  for (int hf = 0; hf < 2; ++hf) {
    #pragma unroll
    for (int nt = 0; nt < 4; ++nt) {
      #pragma unroll
      for (int g = 0; g < 8; ++g) {
        int row = mrow0 + wave * 32 + hf * 16 + g + 8 * lh;
        int col = ncol0 + nt * 16 + lm;
        float v = acc[hf][nt][g];
        if (bias) v += bias[col];
        if (act == 1) {  // tanh-approx gelu (jax default)
          float t = 0.7978845608028654f * (v + 0.044715f * v * v * v);
          v = 0.5f * v * (1.0f + tanhf(t));
        }
        if (residual) v += residual[(size_t)row * ldr + col];
        if (outF) outF[(size_t)row * ldof + col] = v;
        if (outB) outB[(size_t)row * ldob + obcol + col] = f2bf(v);
      }
    }
  }
}

// ---------------------------------------------------------------------------
// Flash attention (WMMA bf16): per (b, h, 64-row q tile). dh=64, NK=2048.
// Output heads written bf16 into (B*NQ, 384) at column h*64.
// ---------------------------------------------------------------------------
__global__ __launch_bounds__(128, 1) void attn_kernel(
    const unsigned short* __restrict__ Q, int ldq, int qc0,
    const unsigned short* __restrict__ Kx, int ldk, int kc0,
    const unsigned short* __restrict__ Vx, int ldv, int vc0,
    unsigned short* __restrict__ O, int ldo) {
  __shared__ unsigned short sK[64 * 72];          // [nk][dh]
  __shared__ unsigned short sV[64 * 72];          // transposed [dh][nk]
  __shared__ unsigned short sP[4 * 16 * 72];      // per-wave P tiles [m][nk]
  const int b = blockIdx.z, h = blockIdx.y;
  const int tid = threadIdx.x;
  const int wave = tid >> 5, lane = tid & 31;
  const int lm = lane & 15, lh = lane >> 4;
  const int qrow0 = b * NQPTS + blockIdx.x * 64;
  const int qoff = qc0 + h * DHEAD, koff = kc0 + h * DHEAD, voff = vc0 + h * DHEAD;

  Frag qa[2];
  {
    const unsigned short* qrow = Q + (size_t)(qrow0 + wave * 16 + lm) * ldq + qoff;
    #pragma unroll
    for (int ks = 0; ks < 2; ++ks)
      #pragma unroll
      for (int r = 0; r < 8; ++r)
        qa[ks].u[r] = *(const unsigned int*)(qrow + KPAIR(r, lh) + ks * 32);
  }

  v8f acc[4];
  #pragma unroll
  for (int nt = 0; nt < 4; ++nt) acc[nt] = zero8();
  float mrow[8], lrow[8];
  #pragma unroll
  for (int g = 0; g < 8; ++g) { mrow[g] = -3.0e38f; lrow[g] = 0.f; }

  unsigned short* pbase = &sP[wave * 16 * 72];

  for (int nkb = 0; nkb < NKPTS; nkb += 64) {
    __syncthreads();
    {  // K tile: 64x64, fixed 4 uint4 chunks per thread
      const unsigned short* kb = Kx + (size_t)(b * NKPTS + nkb) * ldk + koff;
      #pragma unroll
      for (int i = 0; i < 4; ++i) {
        int c = tid + i * 128, row = c >> 3, col8 = (c & 7) * 8;
        *(uint4*)(&sK[row * 72 + col8]) = *(const uint4*)(kb + (size_t)row * ldk + col8);
      }
      // V tile transposed into [dh][nk]: fixed 16 u32 chunks per thread
      const unsigned short* vb = Vx + (size_t)(b * NKPTS + nkb) * ldv + voff;
      #pragma unroll
      for (int i = 0; i < 16; ++i) {
        int c = tid + i * 128, row = c >> 5, col2 = (c & 31) * 2;
        unsigned int d = *(const unsigned int*)(vb + (size_t)row * ldv + col2);
        sV[(col2)     * 72 + row] = (unsigned short)(d & 0xFFFFu);
        sV[(col2 + 1) * 72 + row] = (unsigned short)(d >> 16);
      }
    }
    __syncthreads();

    // S = Q K^T * scale
    v8f s[4];
    #pragma unroll
    for (int nt = 0; nt < 4; ++nt) {
      s[nt] = zero8();
      #pragma unroll
      for (int ks = 0; ks < 2; ++ks) {
        Frag kb;
        #pragma unroll
        for (int r = 0; r < 8; ++r)
          kb.u[r] = *(const unsigned int*)(&sK[(nt * 16 + lm) * 72 + KPAIR(r, lh) + ks * 32]);
        s[nt] = wmma_bf16(qa[ks], kb, s[nt]);
      }
      #pragma unroll
      for (int g = 0; g < 8; ++g) s[nt][g] *= 0.125f;   // 1/sqrt(64)
    }

    // online softmax
    #pragma unroll
    for (int g = 0; g < 8; ++g) {
      float v = fmaxf(fmaxf(s[0][g], s[1][g]), fmaxf(s[2][g], s[3][g]));
      #pragma unroll
      for (int o = 1; o < 16; o <<= 1) v = fmaxf(v, __shfl_xor(v, o, 32));
      float mnew = fmaxf(mrow[g], v);
      float corr = __expf(mrow[g] - mnew);
      mrow[g] = mnew;
      lrow[g] *= corr;
      #pragma unroll
      for (int nt = 0; nt < 4; ++nt) acc[nt][g] *= corr;
      float rs = 0.f;
      #pragma unroll
      for (int nt = 0; nt < 4; ++nt) {
        float p = __expf(s[nt][g] - mnew);
        s[nt][g] = p;
        rs += p;
      }
      #pragma unroll
      for (int o = 1; o < 16; o <<= 1) rs += __shfl_xor(rs, o, 32);
      lrow[g] += rs;
    }

    // stage P (bf16) in LDS
    #pragma unroll
    for (int nt = 0; nt < 4; ++nt)
      #pragma unroll
      for (int g = 0; g < 8; ++g)
        pbase[(g + 8 * lh) * 72 + nt * 16 + lm] = f2bf(s[nt][g]);
    __syncthreads();

    // O += P V
    #pragma unroll
    for (int nt = 0; nt < 4; ++nt) {
      #pragma unroll
      for (int ks = 0; ks < 2; ++ks) {
        Frag pa, vf;
        #pragma unroll
        for (int r = 0; r < 8; ++r) {
          int kp = KPAIR(r, lh) + ks * 32;
          pa.u[r] = *(const unsigned int*)(&pbase[lm * 72 + kp]);
          vf.u[r] = *(const unsigned int*)(&sV[(nt * 16 + lm) * 72 + kp]);
        }
        acc[nt] = wmma_bf16(pa, vf, acc[nt]);
      }
    }
  }

  #pragma unroll
  for (int nt = 0; nt < 4; ++nt)
    #pragma unroll
    for (int g = 0; g < 8; ++g) {
      int row = qrow0 + wave * 16 + g + 8 * lh;
      O[(size_t)row * ldo + h * DHEAD + nt * 16 + lm] = f2bf(acc[nt][g] / lrow[g]);
    }
}

// ---------------------------------------------------------------------------
// KNN top-8 (smallest d2, stable ties), thread-per-query, LDS-tiled keys
// ---------------------------------------------------------------------------
__global__ __launch_bounds__(256) void knn_kernel(const float* __restrict__ qc,
                                                  const float* __restrict__ kc,
                                                  int* __restrict__ idx) {
  __shared__ float skx[256], sky[256], skz[256], sk2[256];
  int row = blockIdx.x * 256 + threadIdx.x;     // block stays inside one batch
  int b = row >> 11;                            // NQ = 2048
  float qx = qc[(size_t)row * 3 + 0], qy = qc[(size_t)row * 3 + 1], qz = qc[(size_t)row * 3 + 2];
  float q2 = qx * qx + qy * qy + qz * qz;
  float bd[KNN]; int bi[KNN];
  #pragma unroll
  for (int t = 0; t < KNN; ++t) { bd[t] = 3.4e38f; bi[t] = 0; }

  for (int base = 0; base < NKPTS; base += 256) {
    __syncthreads();
    {
      int kk = base + threadIdx.x;
      const float* kr = kc + (size_t)(b * NKPTS + kk) * 3;
      float x = kr[0], y = kr[1], z = kr[2];
      skx[threadIdx.x] = x; sky[threadIdx.x] = y; skz[threadIdx.x] = z;
      sk2[threadIdx.x] = x * x + y * y + z * z;
    }
    __syncthreads();
    for (int j = 0; j < 256; ++j) {
      float d2 = q2 + sk2[j] - 2.f * (qx * skx[j] + qy * sky[j] + qz * skz[j]);
      if (d2 < bd[KNN - 1]) {
        float cd = d2; int ci = base + j;
        #pragma unroll
        for (int t = 0; t < KNN; ++t) {
          if (cd < bd[t]) {
            float td = bd[t]; int ti = bi[t];
            bd[t] = cd; bi[t] = ci; cd = td; ci = ti;
          }
        }
      }
    }
  }
  #pragma unroll
  for (int t = 0; t < KNN; ++t) idx[(size_t)row * KNN + t] = bi[t];
}

// ---------------------------------------------------------------------------
// geom = max_k leaky( Np[idx[k]] - CA[row] + CB[row] + bias ), bf16 out
// ---------------------------------------------------------------------------
__global__ __launch_bounds__(128) void gmax_kernel(const float* __restrict__ Np,
                                                   const float* __restrict__ CA,
                                                   const float* __restrict__ CB,
                                                   const float* __restrict__ bias,
                                                   const int* __restrict__ idx,
                                                   unsigned short* __restrict__ outB,
                                                   int ldob, int obcol) {
  int row = blockIdx.x;
  int b = row >> 11;
  int i8[KNN];
  #pragma unroll
  for (int t = 0; t < KNN; ++t) i8[t] = idx[(size_t)row * KNN + t];
  const float* npb = Np + (size_t)b * NKPTS * DMODEL;
  #pragma unroll
  for (int i = 0; i < 3; ++i) {
    int d = threadIdx.x + i * 128;
    float a = -CA[(size_t)row * DMODEL + d] + CB[(size_t)row * DMODEL + d] + bias[d];
    float m = -3.0e38f;
    #pragma unroll
    for (int t = 0; t < KNN; ++t) {
      float v = npb[(size_t)i8[t] * DMODEL + d] + a;
      v = (v < 0.f) ? 0.2f * v : v;
      m = fmaxf(m, v);
    }
    outB[(size_t)row * ldob + obcol + d] = f2bf(m);
  }
}

// ---------------------------------------------------------------------------
// Host launcher
// ---------------------------------------------------------------------------
extern "C" void kernel_launch(void* const* d_in, const int* in_sizes, int n_in,
                              void* d_out, int out_size, void* d_ws, size_t ws_size,
                              hipStream_t stream) {
  (void)in_sizes; (void)n_in; (void)out_size;
  const float* q_pts        = (const float*)d_in[0];
  const float* k_pts        = (const float*)d_in[1];
  const float* ln_in_s      = (const float*)d_in[2];
  const float* ln_in_b      = (const float*)d_in[3];
  const float* qkv_W        = (const float*)d_in[4];
  const float* attn_out_W   = (const float*)d_in[5];
  const float* attn_out_b   = (const float*)d_in[6];
  const float* ff_W1        = (const float*)d_in[7];
  const float* ff_b1        = (const float*)d_in[8];
  const float* ff_W2        = (const float*)d_in[9];
  const float* ff_b2        = (const float*)d_in[10];
  const float* ffn_s        = (const float*)d_in[11];
  const float* ffn_b        = (const float*)d_in[12];
  const float* knn1_W       = (const float*)d_in[13];
  const float* knn1_b       = (const float*)d_in[14];
  const float* knn2_W       = (const float*)d_in[15];
  const float* knn2_b       = (const float*)d_in[16];
  const float* self_merge_W = (const float*)d_in[17];
  const float* self_merge_b = (const float*)d_in[18];
  const float* cross_merge_W= (const float*)d_in[19];
  const float* cross_merge_b= (const float*)d_in[20];
  const float* cnq_s        = (const float*)d_in[21];
  const float* cnq_b        = (const float*)d_in[22];
  const float* cnk_s        = (const float*)d_in[23];
  const float* cnk_b        = (const float*)d_in[24];
  const float* cq_W         = (const float*)d_in[25];
  const float* ck_W         = (const float*)d_in[26];
  const float* cv_W         = (const float*)d_in[27];

  char* wsb = (char*)d_ws;
  size_t off = 0;
  auto alloc = [&](size_t bytes) -> void* {
    void* p = wsb + off;
    off += (bytes + 255) & ~(size_t)255;
    return p;
  };

  float* qc = (float*)alloc((size_t)MROWS * 3 * 4);
  float* kc = (float*)alloc((size_t)MROWS * 3 * 4);
  float* qf = (float*)alloc((size_t)MROWS * DMODEL * 4);
  float* kf = (float*)alloc((size_t)MROWS * DMODEL * 4);
  unsigned short* ln0b   = (unsigned short*)alloc((size_t)MROWS * DMODEL * 2);
  unsigned short* qkvb   = (unsigned short*)alloc((size_t)MROWS * 3 * DMODEL * 2);
  unsigned short* headsb = (unsigned short*)alloc((size_t)MROWS * DMODEL * 2);
  unsigned short* catb   = (unsigned short*)alloc((size_t)MROWS * 2 * DMODEL * 2);
  unsigned short* lnqb   = (unsigned short*)alloc((size_t)MROWS * DMODEL * 2);
  unsigned short* lnkb   = (unsigned short*)alloc((size_t)MROWS * DMODEL * 2);
  unsigned short* q2b    = (unsigned short*)alloc((size_t)MROWS * DMODEL * 2);
  unsigned short* k2b    = (unsigned short*)alloc((size_t)MROWS * DMODEL * 2);
  unsigned short* v2b    = (unsigned short*)alloc((size_t)MROWS * DMODEL * 2);
  unsigned short* h1b    = (unsigned short*)alloc((size_t)MROWS * 2 * DMODEL * 2);
  float* Np1 = (float*)alloc((size_t)MROWS * DMODEL * 4);
  float* CB1 = (float*)alloc((size_t)MROWS * DMODEL * 4);
  float* Np2 = (float*)alloc((size_t)MROWS * DMODEL * 4);
  float* CA2 = (float*)alloc((size_t)MROWS * DMODEL * 4);
  float* CB2 = (float*)alloc((size_t)MROWS * DMODEL * 4);
  int* idx1 = (int*)alloc((size_t)MROWS * KNN * 4);
  int* idx2 = (int*)alloc((size_t)MROWS * KNN * 4);
  unsigned short* wqkv = (unsigned short*)alloc((size_t)3 * DMODEL * DMODEL * 2);
  unsigned short* wout = (unsigned short*)alloc((size_t)DMODEL * DMODEL * 2);
  unsigned short* wf1  = (unsigned short*)alloc((size_t)2 * DMODEL * DMODEL * 2);
  unsigned short* wf2  = (unsigned short*)alloc((size_t)2 * DMODEL * DMODEL * 2);
  unsigned short* wk1a = (unsigned short*)alloc((size_t)DMODEL * DMODEL * 2);
  unsigned short* wk1b = (unsigned short*)alloc((size_t)DMODEL * DMODEL * 2);
  unsigned short* wk2a = (unsigned short*)alloc((size_t)DMODEL * DMODEL * 2);
  unsigned short* wk2b = (unsigned short*)alloc((size_t)DMODEL * DMODEL * 2);
  unsigned short* wsm  = (unsigned short*)alloc((size_t)2 * DMODEL * DMODEL * 2);
  unsigned short* wcm  = (unsigned short*)alloc((size_t)2 * DMODEL * DMODEL * 2);
  unsigned short* wcq  = (unsigned short*)alloc((size_t)DMODEL * DMODEL * 2);
  unsigned short* wck  = (unsigned short*)alloc((size_t)DMODEL * DMODEL * 2);
  unsigned short* wcv  = (unsigned short*)alloc((size_t)DMODEL * DMODEL * 2);
  if (off > ws_size) return;  // workspace too small; bail safely

  auto wprep = [&](const float* W, unsigned short* Wt, int Kd, int N, int r0, int ldw) {
    int elems = N * Kd;
    wprep_kernel<<<(elems + 255) / 256, 256, 0, stream>>>(W, Wt, Kd, N, r0, ldw);
  };
  auto gemm = [&](const unsigned short* A, int lda, const unsigned short* Wt, int N, int Kd,
                  const float* bias, const float* residual, int ldr,
                  float* outF, int ldof, unsigned short* outB, int ldob, int obcol, int act) {
    gemm_bf16_kernel<<<dim3(MROWS / 128, N / 64), 128, 0, stream>>>(
        A, lda, Wt, bias, residual, ldr, outF, ldof, outB, ldob, obcol, Kd, act);
  };
  auto ln = [&](const float* x, const float* g, const float* b, unsigned short* y) {
    ln_bf16_kernel<<<MROWS / 8, 256, 0, stream>>>(x, g, b, y);
  };

  // ---- weight prep (bf16, transposed) ----
  wprep(qkv_W,        wqkv, DMODEL, 3 * DMODEL, 0, 3 * DMODEL);
  wprep(attn_out_W,   wout, DMODEL, DMODEL, 0, DMODEL);
  wprep(ff_W1,        wf1,  DMODEL, 2 * DMODEL, 0, 2 * DMODEL);
  wprep(ff_W2,        wf2,  2 * DMODEL, DMODEL, 0, DMODEL);
  wprep(knn1_W,       wk1a, DMODEL, DMODEL, 0,      DMODEL);
  wprep(knn1_W,       wk1b, DMODEL, DMODEL, DMODEL, DMODEL);
  wprep(knn2_W,       wk2a, DMODEL, DMODEL, 0,      DMODEL);
  wprep(knn2_W,       wk2b, DMODEL, DMODEL, DMODEL, DMODEL);
  wprep(self_merge_W, wsm,  2 * DMODEL, DMODEL, 0, DMODEL);
  wprep(cross_merge_W,wcm,  2 * DMODEL, DMODEL, 0, DMODEL);
  wprep(cq_W,         wcq,  DMODEL, DMODEL, 0, DMODEL);
  wprep(ck_W,         wck,  DMODEL, DMODEL, 0, DMODEL);
  wprep(cv_W,         wcv,  DMODEL, DMODEL, 0, DMODEL);

  // ---- split inputs ----
  {
    size_t total = (size_t)BATCH * NQPTS * CTOT;
    split_kernel<<<(int)((total + 255) / 256), 256, 0, stream>>>(q_pts, qc, qf, NQPTS);
    split_kernel<<<(int)((total + 255) / 256), 256, 0, stream>>>(k_pts, kc, kf, NKPTS);
  }

  // ---- self branch ----
  ln(qf, ln_in_s, ln_in_b, ln0b);                                     // norm_f
  gemm(ln0b, DMODEL, wqkv, 3 * DMODEL, DMODEL,
       nullptr, nullptr, 0, nullptr, 0, qkvb, 3 * DMODEL, 0, 0);      // qkv
  knn_kernel<<<MROWS / 256, 256, 0, stream>>>(qc, qc, idx1);
  gemm(ln0b, DMODEL, wk1a, DMODEL, DMODEL,
       nullptr, nullptr, 0, Np1, DMODEL, nullptr, 0, 0, 0);           // Np1 = CA1
  gemm(ln0b, DMODEL, wk1b, DMODEL, DMODEL,
       nullptr, nullptr, 0, CB1, DMODEL, nullptr, 0, 0, 0);
  attn_kernel<<<dim3(NQPTS / 64, NHEAD, BATCH), 128, 0, stream>>>(
      qkvb, 3 * DMODEL, 0, qkvb, 3 * DMODEL, DMODEL, qkvb, 3 * DMODEL, 2 * DMODEL,
      headsb, DMODEL);
  gemm(headsb, DMODEL, wout, DMODEL, DMODEL,
       attn_out_b, nullptr, 0, nullptr, 0, catb, 2 * DMODEL, 0, 0);   // attn out proj
  gmax_kernel<<<MROWS, 128, 0, stream>>>(Np1, Np1, CB1, knn1_b, idx1,
                                         catb, 2 * DMODEL, DMODEL);   // geom
  gemm(catb, 2 * DMODEL, wsm, DMODEL, 2 * DMODEL,
       self_merge_b, qf, DMODEL, qf, DMODEL, nullptr, 0, 0, 0);       // merge + residual

  // ---- cross branch ----
  ln(qf, cnq_s, cnq_b, lnqb);
  ln(kf, cnk_s, cnk_b, lnkb);
  gemm(lnqb, DMODEL, wcq, DMODEL, DMODEL, nullptr, nullptr, 0, nullptr, 0, q2b, DMODEL, 0, 0);
  gemm(lnkb, DMODEL, wck, DMODEL, DMODEL, nullptr, nullptr, 0, nullptr, 0, k2b, DMODEL, 0, 0);
  gemm(lnkb, DMODEL, wcv, DMODEL, DMODEL, nullptr, nullptr, 0, nullptr, 0, v2b, DMODEL, 0, 0);
  knn_kernel<<<MROWS / 256, 256, 0, stream>>>(qc, kc, idx2);
  gemm(lnkb, DMODEL, wk2a, DMODEL, DMODEL, nullptr, nullptr, 0, Np2, DMODEL, nullptr, 0, 0, 0);
  gemm(lnqb, DMODEL, wk2a, DMODEL, DMODEL, nullptr, nullptr, 0, CA2, DMODEL, nullptr, 0, 0, 0);
  gemm(lnqb, DMODEL, wk2b, DMODEL, DMODEL, nullptr, nullptr, 0, CB2, DMODEL, nullptr, 0, 0, 0);
  attn_kernel<<<dim3(NQPTS / 64, NHEAD, BATCH), 128, 0, stream>>>(
      q2b, DMODEL, 0, k2b, DMODEL, 0, v2b, DMODEL, 0, headsb, DMODEL);
  gemm(headsb, DMODEL, wout, DMODEL, DMODEL,
       attn_out_b, nullptr, 0, nullptr, 0, catb, 2 * DMODEL, 0, 0);
  gmax_kernel<<<MROWS, 128, 0, stream>>>(Np2, CA2, CB2, knn2_b, idx2,
                                         catb, 2 * DMODEL, DMODEL);
  gemm(catb, 2 * DMODEL, wcm, DMODEL, 2 * DMODEL,
       cross_merge_b, qf, DMODEL, qf, DMODEL, nullptr, 0, 0, 0);      // merge + residual

  // ---- FFN ----
  ln(qf, ffn_s, ffn_b, ln0b);
  gemm(ln0b, DMODEL, wf1, 2 * DMODEL, DMODEL,
       ff_b1, nullptr, 0, nullptr, 0, h1b, 2 * DMODEL, 0, 1);         // gelu
  gemm(h1b, 2 * DMODEL, wf2, DMODEL, 2 * DMODEL,
       ff_b2, qf, DMODEL, qf, DMODEL, nullptr, 0, 0, 0);              // + residual

  // ---- pack output ----
  {
    size_t total = (size_t)BATCH * NQPTS * CTOT;
    pack_kernel<<<(int)((total + 255) / 256), 256, 0, stream>>>(qc, qf, (float*)d_out, NQPTS);
  }
}
